// LightningIndexer_34840774705578
// MI455X (gfx1250) — matrix operand order repaired and checked by reading, not compiled, not run on hardware
//
#include <hip/hip_runtime.h>

typedef __attribute__((ext_vector_type(16))) __bf16 v16bf;
typedef __attribute__((ext_vector_type(8)))  __bf16 v8bf;
typedef __attribute__((ext_vector_type(8)))  float  v8f;

#define B_   4
#define T_   4096
#define E_   768
#define H_   6
#define D_   64
#define HD_  384
#define N1_  1152      // 3 * H * D  (q | k | g concatenated)
#define G_   1024      // T / RATIO
#define BT_  (B_ * T_)
#define TOPK_ 8
#define SCALE_ (1.0f / 48.0f)   // (1/H) * D^-0.5 = 1/6 * 1/8

__device__ __forceinline__ unsigned short f2bf(float f) {
  unsigned int u = __float_as_uint(f);
  u += 0x7FFFu + ((u >> 16) & 1u);          // round-to-nearest-even
  return (unsigned short)(u >> 16);
}
__device__ __forceinline__ float bf2f(unsigned short h) {
  return __uint_as_float(((unsigned int)h) << 16);
}
// A fragment (16x32 bf16): lanes 0-15 hold K {kk..kk+15},
// lanes 16-31 hold K {kk+16..kk+31}  -> two 16B loads.
__device__ __forceinline__ v16bf load_afrag(const unsigned short* p) {
  v8bf lo = *reinterpret_cast<const v8bf*>(p);
  v8bf hi = *reinterpret_cast<const v8bf*>(p + 8);
  return __builtin_shufflevector(lo, hi, 0,1,2,3,4,5,6,7,8,9,10,11,12,13,14,15);
}
__device__ __forceinline__ float wave_sum(float v) {
  #pragma unroll
  for (int m = 16; m >= 1; m >>= 1) v += __shfl_xor(v, m, 32);
  return v;
}

// ---------------- Stage 0a: x f32 -> bf16 ----------------
__global__ void k_cvt_x(const float* __restrict__ x, unsigned short* __restrict__ xb) {
  int i = blockIdx.x * blockDim.x + threadIdx.x;
  if (i >= (BT_ * E_) / 4) return;
  float4 v = reinterpret_cast<const float4*>(x)[i];
  ushort4 o;
  o.x = f2bf(v.x); o.y = f2bf(v.y); o.z = f2bf(v.z); o.w = f2bf(v.w);
  reinterpret_cast<ushort4*>(xb)[i] = o;
}

// ---------------- Stage 0b: build wT[1152][768] bf16 (transposed, q|k|g) ----
__global__ void k_build_wT(const float* __restrict__ wq, const float* __restrict__ wk,
                           const float* __restrict__ wg, unsigned short* __restrict__ wT) {
  int id = blockIdx.x * blockDim.x + threadIdx.x;
  if (id >= N1_ * E_) return;
  int n = id / E_, k = id % E_;
  int sel = n / HD_, c = n % HD_;
  const float* W = (sel == 0) ? wq : (sel == 1) ? wk : wg;
  wT[(size_t)n * E_ + k] = f2bf(W[(size_t)k * HD_ + c]);
}

// ---------------- Stage 1: qkg[16384][1152] = x @ [wq|wk|wg]  (bf16 WMMA) ----
// Each wave: 32 tokens x 64 cols (2 M-tiles share every B fragment).
__global__ void k_gemm1(const unsigned short* __restrict__ xb,
                        const unsigned short* __restrict__ wT,
                        unsigned short* __restrict__ qkg) {
  const int tid  = threadIdx.x;
  const int wave = tid >> 5, lane = tid & 31;
  const int mt2 = blockIdx.y * 4 + wave;         // 0..511
  const int t0 = mt2 * 32;
  const int n0 = blockIdx.x * 64;
  const int l16 = lane & 15, lh = lane >> 4;

  const unsigned short* arow0 = xb + (size_t)(t0 + l16)      * E_ + lh * 16;
  const unsigned short* arow1 = xb + (size_t)(t0 + 16 + l16) * E_ + lh * 16;
  v8f acc0[4] = {{}, {}, {}, {}};
  v8f acc1[4] = {{}, {}, {}, {}};
  for (int kk = 0; kk < E_; kk += 32) {
    v16bf a0 = load_afrag(arow0 + kk);
    v16bf a1 = load_afrag(arow1 + kk);
    #pragma unroll
    for (int j = 0; j < 4; ++j) {
      const unsigned short* bp =
          wT + (size_t)(n0 + j * 16 + l16) * E_ + kk + lh * 16;
      v16bf b = *reinterpret_cast<const v16bf*>(bp);
      acc0[j] = __builtin_amdgcn_wmma_f32_16x16x32_bf16(
          false, a0, false, b, (short)0, acc0[j], false, false);
      acc1[j] = __builtin_amdgcn_wmma_f32_16x16x32_bf16(
          false, a1, false, b, (short)0, acc1[j], false, false);
    }
  }
  // C layout: elem i -> row base + i + 8*(lane>>4), col n0 + j*16 + (lane&15)
  #pragma unroll
  for (int j = 0; j < 4; ++j)
    #pragma unroll
    for (int i = 0; i < 8; ++i) {
      qkg[(size_t)(t0 + i + lh * 8) * N1_      + n0 + j * 16 + l16] = f2bf(acc0[j][i]);
      qkg[(size_t)(t0 + 16 + i + lh * 8) * N1_ + n0 + j * 16 + l16] = f2bf(acc1[j][i]);
    }
}

// ---------------- Stage 2a: RMS-norm q in place (per token, per head) -------
__global__ void k_qnorm(unsigned short* __restrict__ qkg) {
  const int wave = threadIdx.x >> 5, lane = threadIdx.x & 31;
  const int gid = blockIdx.x * 4 + wave;          // 0..98303
  const int t = gid / H_, h = gid % H_;
  unsigned short* p = qkg + (size_t)t * N1_ + h * D_ + lane * 2;
  unsigned int u = *reinterpret_cast<unsigned int*>(p);
  float f0 = bf2f((unsigned short)(u & 0xFFFF));
  float f1 = bf2f((unsigned short)(u >> 16));
  float ss = wave_sum(f0 * f0 + f1 * f1);
  float r = rsqrtf(ss * (1.0f / D_) + 1e-6f);
  unsigned int o = (unsigned int)f2bf(f0 * r) | ((unsigned int)f2bf(f1 * r) << 16);
  *reinterpret_cast<unsigned int*>(p) = o;
}

// ---------------- Stage 2b: gated pooling + RMS-norm -> keys[B][G][384] -----
__global__ void k_pool(const unsigned short* __restrict__ qkg,
                       const float* __restrict__ ape,
                       unsigned short* __restrict__ keysb) {
  const int wave = threadIdx.x >> 5, lane = threadIdx.x & 31;
  const int gid = blockIdx.x * 4 + wave;          // 0..24575
  const int h = gid % H_;
  const int g = (gid / H_) % G_;
  const int b = gid / (H_ * G_);
  const int d0 = lane * 2;

  float kv[4][2], gv[4][2];
  #pragma unroll
  for (int r = 0; r < 4; ++r) {
    size_t row = (size_t)(b * T_ + g * 4 + r) * N1_;
    unsigned int uk = *reinterpret_cast<const unsigned int*>(qkg + row + HD_     + h * D_ + d0);
    unsigned int ug = *reinterpret_cast<const unsigned int*>(qkg + row + 2 * HD_ + h * D_ + d0);
    float2 ap = *reinterpret_cast<const float2*>(ape + (size_t)(r * H_ + h) * D_ + d0);
    kv[r][0] = bf2f((unsigned short)(uk & 0xFFFF));
    kv[r][1] = bf2f((unsigned short)(uk >> 16));
    gv[r][0] = bf2f((unsigned short)(ug & 0xFFFF)) + ap.x;
    gv[r][1] = bf2f((unsigned short)(ug >> 16))   + ap.y;
  }
  float key[2];
  #pragma unroll
  for (int x = 0; x < 2; ++x) {
    float m = fmaxf(fmaxf(gv[0][x], gv[1][x]), fmaxf(gv[2][x], gv[3][x]));
    float s = 0.0f, acc = 0.0f;
    #pragma unroll
    for (int r = 0; r < 4; ++r) {
      float e = __expf(gv[r][x] - m);
      s += e; acc += kv[r][x] * e;
    }
    key[x] = acc / s;
  }
  float ss = wave_sum(key[0] * key[0] + key[1] * key[1]);
  float r = rsqrtf(ss * (1.0f / D_) + 1e-6f);
  unsigned int o = (unsigned int)f2bf(key[0] * r) | ((unsigned int)f2bf(key[1] * r) << 16);
  *reinterpret_cast<unsigned int*>(keysb + (size_t)(b * G_ + g) * HD_ + h * D_ + d0) = o;
}

// ---------------- Stage 3: fused score GEMM (bf16 WMMA) + causal top-8 ------
__global__ void k_scores_topk(const unsigned short* __restrict__ qkg,
                              const unsigned short* __restrict__ keysb,
                              float* __restrict__ out) {
  __shared__ float sT[4][16][17];
  const int wave = threadIdx.x >> 5, lane = threadIdx.x & 31;
  const int l16 = lane & 15, lh = lane >> 4;
  const int b = blockIdx.y;
  const int ttile = blockIdx.x * 4 + wave;        // 0..255
  const int t0 = ttile * 16;

  // Preload all 12 A fragments (q rows stay resident across the whole g loop)
  const unsigned short* arow = qkg + (size_t)(b * T_ + t0 + l16) * N1_ + lh * 16;
  v16bf afr[12];
  #pragma unroll
  for (int k = 0; k < 12; ++k) afr[k] = load_afrag(arow + k * 32);

  float tv[TOPK_];
  int   ti[TOPK_];
  #pragma unroll
  for (int q = 0; q < TOPK_; ++q) { tv[q] = -__builtin_inff(); ti[q] = -1; }

  const int tmax = t0 + 15;
  const int ntile = ((tmax - 3) >> 2) >> 4;       // last causal g-tile index

  for (int gt = 0; gt <= ntile; ++gt) {
    const int g0 = gt * 16;
    const unsigned short* bbase =
        keysb + (size_t)(b * G_ + g0 + l16) * HD_ + lh * 16;
    __builtin_prefetch(bbase + 16 * HD_, 0, 1);   // next tile of keys
    v8f acc = {};
    #pragma unroll
    for (int k = 0; k < 12; ++k) {
      v16bf bf = *reinterpret_cast<const v16bf*>(bbase + k * 32);
      acc = __builtin_amdgcn_wmma_f32_16x16x32_bf16(
          false, afr[k], false, bf, (short)0, acc, false, false);
    }
    #pragma unroll
    for (int i = 0; i < 8; ++i)
      sT[wave][i + lh * 8][l16] = acc[i] * SCALE_;
    // Same-wave DS ops are in-order: the loads below see the stores above.
    if (lane < 16) {
      const int t = t0 + lane;
      #pragma unroll
      for (int j = 0; j < 16; ++j) {
        const int g = g0 + j;
        if (4 * g + 3 <= t) {
          float v = sT[wave][lane][j];
          float mn = tv[0]; int mi = 0;
          #pragma unroll
          for (int q = 1; q < TOPK_; ++q)
            if (tv[q] < mn) { mn = tv[q]; mi = q; }
          if (v > mn) {
            #pragma unroll
            for (int q = 0; q < TOPK_; ++q)
              if (q == mi) { tv[q] = v; ti[q] = g; }
          }
        }
      }
    }
  }

  // Write mask rows: zeros first, then scatter 1.0 (per-wave stores ordered).
  float4 z4 = {0.f, 0.f, 0.f, 0.f};
  float4* orow4 = reinterpret_cast<float4*>(out + ((size_t)b * T_ + t0) * G_);
  for (int i = 0; i < 16; ++i)
    for (int c = lane; c < G_ / 4; c += 32)
      orow4[i * (G_ / 4) + c] = z4;
  if (lane < 16) {
    float* orow = out + ((size_t)b * T_ + t0 + lane) * G_;
    #pragma unroll
    for (int q = 0; q < TOPK_; ++q)
      if (ti[q] >= 0) orow[ti[q]] = 1.0f;
  }
}

// ---------------- Stage 4: group_ends (as float, after mask region) ---------
__global__ void k_gends(float* __restrict__ out) {
  int i = blockIdx.x * blockDim.x + threadIdx.x;
  if (i < G_) {
    int ge = i * 4 + 3;
    if (ge > T_ - 1) ge = T_ - 1;
    out[(size_t)B_ * T_ * G_ + i] = (float)ge;
  }
}

extern "C" void kernel_launch(void* const* d_in, const int* in_sizes, int n_in,
                              void* d_out, int out_size, void* d_ws, size_t ws_size,
                              hipStream_t stream) {
  const float* x   = (const float*)d_in[0];
  const float* wq  = (const float*)d_in[1];
  const float* wk  = (const float*)d_in[2];
  const float* wg  = (const float*)d_in[3];
  const float* ape = (const float*)d_in[4];
  float* out = (float*)d_out;

  unsigned short* ws    = (unsigned short*)d_ws;
  unsigned short* xb    = ws;                                   // [16384][768]
  unsigned short* wT    = xb  + (size_t)BT_ * E_;               // [1152][768]
  unsigned short* qkg   = wT  + (size_t)N1_ * E_;               // [16384][1152]
  unsigned short* keysb = qkg + (size_t)BT_ * N1_;              // [4][1024][384]

  k_cvt_x     <<<(BT_ * E_ / 4 + 255) / 256, 256, 0, stream>>>(x, xb);
  k_build_wT  <<<(N1_ * E_ + 255) / 256,     256, 0, stream>>>(wq, wk, wg, wT);
  k_gemm1     <<<dim3(N1_ / 64, BT_ / 128),  128, 0, stream>>>(xb, wT, qkg);
  k_qnorm     <<<(BT_ * H_) / 4,             128, 0, stream>>>(qkg);
  k_pool      <<<(B_ * G_ * H_) / 4,         128, 0, stream>>>(qkg, ape, keysb);
  k_scores_topk<<<dim3(T_ / 16 / 4, B_),     128, 0, stream>>>(qkg, keysb, out);
  k_gends     <<<(G_ + 255) / 256,           256, 0, stream>>>(out);
}
